// SRC_66889820668538
// MI455X (gfx1250) — matrix operand order repaired and checked by reading, not compiled, not run on hardware
//
#include <hip/hip_runtime.h>
#include <cmath>

// ---------------------------------------------------------------------------
// MI455X (gfx1250) implementation.
// Algebraic reduction of the reference: only the q/k path affects the outputs;
// heavy work = three bf16 WMMA GEMMs + an L2-resident gathered softmax.
// A-tiles are staged LDS-side by the Tensor Data Mover (tensor_load_to_lds,
// TENSORcnt-tracked), per CDNA5 ISA ch.8.
// ---------------------------------------------------------------------------

typedef __bf16 bf16;
typedef __bf16 v16bf __attribute__((ext_vector_type(16)));
typedef __bf16 v8bf  __attribute__((ext_vector_type(8)));
typedef float  v8f   __attribute__((ext_vector_type(8)));
typedef unsigned int v4u __attribute__((ext_vector_type(4)));
typedef unsigned int v8u __attribute__((ext_vector_type(8)));

// ---------------- elementwise f32 -> bf16 ----------------
__global__ __launch_bounds__(256) void k_f2bf(const float* __restrict__ x,
                                              bf16* __restrict__ y, int n) {
  int i = blockIdx.x * 256 + threadIdx.x;
  if (i < n) y[i] = (bf16)x[i];
}

// ---------------- y[M] = A[M,K] @ x[K]  (wave32 per row) ----------------
__global__ __launch_bounds__(256) void k_matvec(const float* __restrict__ A,
                                                const float* __restrict__ x,
                                                float* __restrict__ y,
                                                int M, int K) {
  const int lane = threadIdx.x & 31;
  const int row  = blockIdx.x * 8 + (threadIdx.x >> 5);
  if (row >= M) return;
  float s = 0.f;
  for (int k = lane; k < K; k += 32) s += A[(size_t)row * K + k] * x[k];
#pragma unroll
  for (int off = 16; off > 0; off >>= 1) s += __shfl_xor(s, off, 32);
  if (lane == 0) y[row] = s;
}

// ---------------- bf16 WMMA GEMM: C[M,N] = A[M,K] @ B ----------------
// BT=false: B is [K,N] row-major.  BT=true: B is [N,K] row-major (B^T).
// Block = 256 threads = 8 waves (4 along M x 2 along N); wave tile 16x32
// (2 WMMAs, shared A fragment); block tile 64x64; K stepped by 32.
// A tile staged by the Tensor Data Mover (ISA ch.8), B tile by VALU loads.
// A/B fragments use contiguous k = 16*(lane/16)+s in both; D layout per CDNA5 ISA 7.12.2 (wave32).
constexpr int BM = 64, BN = 64, BK = 32;

template<bool BT>
__global__ __launch_bounds__(256) void k_gemm_bf16(const bf16* __restrict__ A,
                                                   const bf16* __restrict__ B,
                                                   float* __restrict__ C,
                                                   int M, int N, int K) {
  __shared__ __attribute__((aligned(64))) bf16 As[BM][BK];
  __shared__ __attribute__((aligned(64))) bf16 Bs[BN][BK];  // stored n-major
  const int tid  = threadIdx.x;
  const int lane = tid & 31;
  const int wave = tid >> 5;
  const int wm   = wave & 3;   // 4 waves along M
  const int wn   = wave >> 2;  // 2 waves along N (32 cols each)
  const int m0 = blockIdx.y * BM;
  const int n0 = blockIdx.x * BN;

  const unsigned int lds_a = (unsigned int)(uintptr_t)(&As[0][0]);

  v8f acc0 = {}, acc1 = {};
  for (int k0 = 0; k0 < K; k0 += BK) {
    // ---- A tile (64x32 bf16) via TDM: one tensor_load_to_lds per block ----
    // D# group0/group1 per ISA 8.3/8.4: 2D tile, data_size=2B,
    // tensor dims = remaining extent so OOB rows/cols read as zero.
    if (wave == 0) {
      const unsigned long long ga =
          (unsigned long long)(uintptr_t)(A + (size_t)m0 * K + k0);
      const unsigned int td0 = (unsigned int)(K - k0);   // tensor_dim0 (k)
      const unsigned int td1 = (unsigned int)(M - m0);   // tensor_dim1 (rows)
      v4u g0;
      g0[0] = 1u;                                        // count=1, flags=0
      g0[1] = lds_a;                                     // lds_addr
      g0[2] = (unsigned int)ga;                          // global_addr[31:0]
      g0[3] = (unsigned int)((ga >> 32) & 0x01ffffffu) | (2u << 30); // type=2
      v8u g1;
      g1[0] = 1u << 16;                                  // data_size=1 (2B)
      g1[1] = (td0 & 0xffffu) << 16;                     // tensor_dim0[15:0]
      g1[2] = (td0 >> 16) | ((td1 & 0xffffu) << 16);     // td0[31:16]|td1[15:0]
      g1[3] = (td1 >> 16) | ((unsigned int)BK << 16);    // td1[31:16]|tile_dim0
      g1[4] = (unsigned int)BM;                          // tile_dim1 (tile_dim2=0)
      g1[5] = (unsigned int)K;                           // tensor_dim0_stride lo
      g1[6] = 0u;
      g1[7] = 0u;
      asm volatile("tensor_load_to_lds %0, %1" :: "s"(g0), "s"(g1) : "memory");
      __builtin_amdgcn_s_wait_tensorcnt(0);
    }

    // ---- stage B tile as Bs[n][k]: 64x32 bf16, 8 per thread, branchless ----
    {
      const int idx = tid * 8;
      const int nn  = idx >> 5;       // 0..63
      const int c   = idx & 31;       // multiple of 8
      const int gn  = n0 + nn;
      const int gnc = (gn < N) ? gn : (N - 1);   // clamp; select zero below
      if (BT) {
        v8bf bv = *(const v8bf*)&B[(size_t)gnc * K + k0 + c];
        v8bf zz = {};
        if (gn >= N) bv = zz;
        *(v8bf*)&Bs[nn][c] = bv;
      } else {
        v8bf bv;
#pragma unroll
        for (int j = 0; j < 8; ++j)
          bv[j] = B[(size_t)(k0 + c + j) * N + gnc];
        v8bf zz = {};
        if (gn >= N) bv = zz;
        *(v8bf*)&Bs[nn][c] = bv;
      }
    }
    __syncthreads();

    // prefetch next K-slab of B (gfx1250 global_prefetch_b8 path)
    if (k0 + BK < K) {
      if (BT) {
        const int gn = n0 + (tid >> 2);
        if (gn < N)
          __builtin_prefetch(&B[(size_t)gn * K + k0 + BK + ((tid & 3) << 3)], 0, 0);
      } else {
        const int r = tid >> 3;            // 0..31 rows of next slab
        const int c = (tid & 7) << 3;      // 0..56
        if (n0 + c < N)
          __builtin_prefetch(&B[(size_t)(k0 + BK + r) * N + n0 + c], 0, 0);
      }
    }

    // ---- A fragment (16x32 bf16, lane -> M, 16 consecutive K) ----
    const int mrow  = wm * 16 + (lane & 15);
    const int khalf = (lane < 16) ? 0 : 16;
    v8bf a0 = *(const v8bf*)&As[mrow][khalf];       // K = khalf .. khalf+7
    v8bf a1 = *(const v8bf*)&As[mrow][8 + khalf];   // K = 8+khalf .. 15+khalf
    v16bf af;
#pragma unroll
    for (int i = 0; i < 8; ++i) { af[i] = a0[i]; af[i + 8] = a1[i]; }

    // ---- two B fragments (32x16 bf16): lane -> N, 16 consecutive K ----
    const int ncol0 = wn * 32 + (lane & 15);
    const int kb    = (lane < 16) ? 0 : 16;
    v16bf b0 = *(const v16bf*)&Bs[ncol0][kb];
    v16bf b1 = *(const v16bf*)&Bs[ncol0 + 16][kb];

    acc0 = __builtin_amdgcn_wmma_f32_16x16x32_bf16(false, af, false, b0,
                                                   (short)0, acc0, false, false);
    acc1 = __builtin_amdgcn_wmma_f32_16x16x32_bf16(false, af, false, b1,
                                                   (short)0, acc1, false, false);
    __syncthreads();
  }

  // ---- store D (16x16 f32 layout: VGPR r -> M = r + 8*(lane>=16)) ----
  const int rbase = m0 + wm * 16 + ((lane < 16) ? 0 : 8);
  const int ccol0 = n0 + wn * 32 + (lane & 15);
  const int ccol1 = ccol0 + 16;
  if (ccol0 < N) {
#pragma unroll
    for (int r = 0; r < 8; ++r)
      if (rbase + r < M) C[(size_t)(rbase + r) * N + ccol0] = acc0[r];
  }
  if (ccol1 < N) {
#pragma unroll
    for (int r = 0; r < 8; ++r)
      if (rbase + r < M) C[(size_t)(rbase + r) * N + ccol1] = acc1[r];
  }
}

// ---------------- fused gathered attention + score ----------------
// score0[b,s] = sum_t softmax_t(G[p[b,s],p[b,t]]*scale) * embVV[p[b,t]]
//             + embU[p[b,s]]
// One wave per (b,s) row; G (4MB) stays L2-resident.
__global__ __launch_bounds__(256) void k_attn_score(const int* __restrict__ path,
                                                    const float* __restrict__ G,
                                                    const float* __restrict__ embU,
                                                    const float* __restrict__ embVV,
                                                    float* __restrict__ score0,
                                                    int B, int S, int V, float scale) {
  const int lane = threadIdx.x & 31;
  const int row  = blockIdx.x * 8 + (threadIdx.x >> 5);
  if (row >= B * S) return;
  const int b   = row / S;
  const int qid = path[row];
  const float* __restrict__ Grow = G + (size_t)qid * V;
  const int*   __restrict__ prow = path + (size_t)b * S;

  float mx = -3.0e38f;
  for (int t = lane; t < S; t += 32) mx = fmaxf(mx, Grow[prow[t]] * scale);
#pragma unroll
  for (int off = 16; off > 0; off >>= 1) mx = fmaxf(mx, __shfl_xor(mx, off, 32));

  float se = 0.f, sw = 0.f;
  for (int t = lane; t < S; t += 32) {
    const int kid = prow[t];
    const float e = __expf(Grow[kid] * scale - mx);
    se += e;
    sw += e * embVV[kid];
  }
#pragma unroll
  for (int off = 16; off > 0; off >>= 1) {
    se += __shfl_xor(se, off, 32);
    sw += __shfl_xor(sw, off, 32);
  }
  if (lane == 0) score0[row] = sw / se + embU[qid];
}

// ---------------- iterative masked argmax + prob (n steps) ----------------
// p_sel = 1/sum_unmasked exp(score - max)  (sel is the argmax; -1e9-masked
// terms underflow to exactly 0 in fp32, matching the reference).
__global__ __launch_bounds__(256) void k_select(const float* __restrict__ score0,
                                                const int* __restrict__ path,
                                                float* __restrict__ out,
                                                int B, int S, int n) {
  __shared__ float sval[256];
  __shared__ int   sidx[256];
  __shared__ float ssum[256];
  const int b = blockIdx.x;
  const int s = threadIdx.x;

  const float sc = (s < S) ? score0[(size_t)b * S + s] : -3.0e38f;
  bool sel = (s >= S);

  float* paths = out;
  float* probs = out + (size_t)B * n;
  float* sels  = out + (size_t)2 * B * n;

  for (int step = 0; step < n; ++step) {
    // argmax over unselected (first-occurrence tie-break, like jnp.argmax)
    sval[s] = sel ? -3.0e38f : sc;
    sidx[s] = s;
    __syncthreads();
#pragma unroll
    for (int off = 128; off > 0; off >>= 1) {
      if (s < off) {
        const float v2 = sval[s + off];
        const int   i2 = sidx[s + off];
        if (v2 > sval[s] || (v2 == sval[s] && i2 < sidx[s])) {
          sval[s] = v2; sidx[s] = i2;
        }
      }
      __syncthreads();
    }
    const float m  = sval[0];
    const int   am = sidx[0];
    __syncthreads();

    ssum[s] = sel ? 0.f : __expf(sc - m);
    __syncthreads();
#pragma unroll
    for (int off = 128; off > 0; off >>= 1) {
      if (s < off) ssum[s] += ssum[s + off];
      __syncthreads();
    }
    if (s == 0) {
      paths[(size_t)b * n + step] = (float)path[(size_t)b * S + am];
      probs[(size_t)b * n + step] = 1.0f / ssum[0];
      sels[(size_t)b * n + step]  = (float)am;
    }
    if (s == am) sel = true;
    __syncthreads();
  }
}

// ---------------------------------------------------------------------------
extern "C" void kernel_launch(void* const* d_in, const int* in_sizes, int n_in,
                              void* d_out, int out_size, void* d_ws, size_t ws_size,
                              hipStream_t stream) {
  const int S = 200, D = 512, H = 512;
  const int B = in_sizes[3] / S;
  const int V = in_sizes[5] / D;
  if (B <= 0 || V <= 0) return;
  const int n = out_size / (3 * B);
  if (n <= 0) return;

  const int*   path = (const int*)d_in[3];
  const float* emb  = (const float*)d_in[5];
  const float* l2W  = (const float*)d_in[8];
  const float* Wq   = (const float*)d_in[13];
  const float* Wk   = (const float*)d_in[14];
  const float* Wv   = (const float*)d_in[15];
  const float* Wo   = (const float*)d_in[16];
  const float* W1   = (const float*)d_in[17];
  const float* vt   = (const float*)d_in[19];

  char* base = (char*)d_ws;
  size_t off = 0;
  auto carve = [&](size_t bytes) -> void* {
    void* p = base + off;
    off = (off + bytes + 255) & ~(size_t)255;
    return p;
  };
  bf16*  embB   = (bf16*) carve((size_t)V * D * 2);
  bf16*  l2WB   = (bf16*) carve((size_t)D * H * 2);
  bf16*  WqB    = (bf16*) carve((size_t)H * H * 2);
  bf16*  WkB    = (bf16*) carve((size_t)H * H * 2);
  float* AqF    = (float*)carve((size_t)D * H * 4);
  float* AkF    = (float*)carve((size_t)D * H * 4);
  bf16*  AqB    = (bf16*) carve((size_t)D * H * 2);
  bf16*  AkB    = (bf16*) carve((size_t)D * H * 2);
  float* embQF  = (float*)carve((size_t)V * H * 4);
  float* embKF  = (float*)carve((size_t)V * H * 4);
  bf16*  embQB  = (bf16*) carve((size_t)V * H * 2);
  bf16*  embKB  = (bf16*) carve((size_t)V * H * 2);
  float* G      = (float*)carve((size_t)V * V * 4);
  float* w1v    = (float*)carve((size_t)H * 4);
  float* wvv    = (float*)carve((size_t)H * 4);
  float* wv2    = (float*)carve((size_t)H * 4);
  float* wu     = (float*)carve((size_t)D * 4);
  float* wvp    = (float*)carve((size_t)D * 4);
  float* embU   = (float*)carve((size_t)V * 4);
  float* embVV  = (float*)carve((size_t)V * 4);
  float* score0 = (float*)carve((size_t)B * S * 4);
  if (off > ws_size) return;

  auto cdiv = [](int a, int b) { return (a + b - 1) / b; };

  // 1) bf16 conversions of GEMM operands
  k_f2bf<<<cdiv(V * D, 256), 256, 0, stream>>>(emb, embB, V * D);
  k_f2bf<<<cdiv(D * H, 256), 256, 0, stream>>>(l2W, l2WB, D * H);
  k_f2bf<<<cdiv(H * H, 256), 256, 0, stream>>>(Wq, WqB, H * H);
  k_f2bf<<<cdiv(H * H, 256), 256, 0, stream>>>(Wk, WkB, H * H);

  // 2) weight-vector chain (fp32 matvecs):
  //    w1v = W1@vt; wvv = Wo@w1v; wv2 = Wv@wvv; wu = l2_W@w1v; wvp = l2_W@wv2
  //    embU = emb@wu; embVV = emb@wvp
  k_matvec<<<cdiv(H, 8), 256, 0, stream>>>(W1,  vt,  w1v, H, H);
  k_matvec<<<cdiv(H, 8), 256, 0, stream>>>(Wo,  w1v, wvv, H, H);
  k_matvec<<<cdiv(H, 8), 256, 0, stream>>>(Wv,  wvv, wv2, H, H);
  k_matvec<<<cdiv(D, 8), 256, 0, stream>>>(l2W, w1v, wu,  D, H);
  k_matvec<<<cdiv(D, 8), 256, 0, stream>>>(l2W, wv2, wvp, D, H);
  k_matvec<<<cdiv(V, 8), 256, 0, stream>>>(emb, wu,  embU,  V, D);
  k_matvec<<<cdiv(V, 8), 256, 0, stream>>>(emb, wvp, embVV, V, D);

  // 3) WMMA GEMMs: Aq = l2_W@Wq, Ak = l2_W@Wk   (D x H, K = H)
  {
    dim3 g(cdiv(H, BN), cdiv(D, BM));
    k_gemm_bf16<false><<<g, 256, 0, stream>>>(l2WB, WqB, AqF, D, H, H);
    k_gemm_bf16<false><<<g, 256, 0, stream>>>(l2WB, WkB, AkF, D, H, H);
  }
  k_f2bf<<<cdiv(D * H, 256), 256, 0, stream>>>(AqF, AqB, D * H);
  k_f2bf<<<cdiv(D * H, 256), 256, 0, stream>>>(AkF, AkB, D * H);

  // 4) embQ = emb@Aq, embK = emb@Ak   (V x H, K = D)
  {
    dim3 g(cdiv(H, BN), cdiv(V, BM));
    k_gemm_bf16<false><<<g, 256, 0, stream>>>(embB, AqB, embQF, V, H, D);
    k_gemm_bf16<false><<<g, 256, 0, stream>>>(embB, AkB, embKF, V, H, D);
  }
  k_f2bf<<<cdiv(V * H, 256), 256, 0, stream>>>(embQF, embQB, V * H);
  k_f2bf<<<cdiv(V * H, 256), 256, 0, stream>>>(embKF, embKB, V * H);

  // 5) Gram matrix G = embQ @ embK^T   (V x V, K = H) — 4MB, L2-resident
  {
    dim3 g(cdiv(V, BN), cdiv(V, BM));
    k_gemm_bf16<true><<<g, 256, 0, stream>>>(embQB, embKB, G, V, V, H);
  }

  // 6) fused gathered softmax -> score0[b,s]
  const float scale = 1.0f / sqrtf((float)H);
  k_attn_score<<<cdiv(B * S, 8), 256, 0, stream>>>(path, G, embU, embVV,
                                                   score0, B, S, V, scale);

  // 7) n-step masked argmax + probs -> d_out (paths | probs | sels as floats)
  k_select<<<B, 256, 0, stream>>>(score0, path, (float*)d_out, B, S, n);
}
